// CausalMultiHeadAttentionRoPE_43911745634980
// MI455X (gfx1250) — matrix operand order repaired and checked by reading, not compiled, not run on hardware
//
#include <hip/hip_runtime.h>
#include <hip/hip_bf16.h>

// Causal MHA + RoPE for gfx1250 (MI455X), all matmuls via v_wmma_f32_16x16x32_bf16.
// B=4, S=2048, D=1024, H=16, AOD=64.

typedef __bf16 bf16;
typedef __attribute__((ext_vector_type(8)))  bf16  v8bf;
typedef __attribute__((ext_vector_type(16))) bf16  v16bf;
typedef __attribute__((ext_vector_type(8)))  float v8f;

#define CAT16(lo, hi) __builtin_shufflevector((lo), (hi), 0,1,2,3,4,5,6,7,8,9,10,11,12,13,14,15)

// DPP16 lane permute (full-rate VALU, no DS round-trip, no counter waits).
// 0xB1 = quad_perm(1,0,3,2)=xor1, 0x4E = quad_perm(2,3,0,1)=xor2,
// 0x141 = row_half_mirror (combines quads within 8), 0x140 = row_mirror (combines 8-halves within 16).
#define DPPF(x, ctrl) \
  __int_as_float(__builtin_amdgcn_update_dpp(0, __float_as_int(x), (ctrl), 0xF, 0xF, true))

__device__ __forceinline__ float redmax16(float v) {
  v = fmaxf(v, DPPF(v, 0xB1));
  v = fmaxf(v, DPPF(v, 0x4E));
  v = fmaxf(v, DPPF(v, 0x141));
  v = fmaxf(v, DPPF(v, 0x140));
  return v;
}
__device__ __forceinline__ float redsum16(float v) {
  v += DPPF(v, 0xB1);
  v += DPPF(v, 0x4E);
  v += DPPF(v, 0x141);
  v += DPPF(v, 0x140);
  return v;
}

__device__ __forceinline__ v8f wmma_bf16(v16bf a, v16bf b, v8f c) {
  // 8 args: (neg_a, A, neg_b, B, c_mod, C, reuse_a, reuse_b)
  return __builtin_amdgcn_wmma_f32_16x16x32_bf16(false, a, false, b, (short)0, c, false, false);
}

// A-fragment 16x32 bf16 from row-major A (lda elems). Lane l: row m0+(l&15);
// elems 0..7 = K[off..off+7], elems 8..15 = K[off+16..off+23], off=(l>=16)*8.
__device__ __forceinline__ v16bf load_a_frag(const bf16* __restrict__ A, int lda,
                                             int m0, int k0, int lane) {
  const int m    = m0 + (lane & 15);
  const int koff = k0 + ((lane >> 4) & 1) * 8;
  const bf16* p  = A + (size_t)m * lda + koff;
  v8bf lo = *reinterpret_cast<const v8bf*>(p);
  v8bf hi = *reinterpret_cast<const v8bf*>(p + 16);
  return CAT16(lo, hi);
}

// B-fragment 32x16 bf16 from K-major storage Bt[n][k] (ldb elems along k).
// Lane l: col n0+(l&15); elems 0..7 = K[off..off+7], elems 8..15 = K[off+16..off+23], off=(l>=16)*8.
__device__ __forceinline__ v16bf load_b_frag(const bf16* __restrict__ Bt, int ldb,
                                             int n0, int k0, int lane) {
  const int n    = n0 + (lane & 15);
  const int koff = k0 + ((lane >> 4) & 1) * 8;
  const bf16* p  = Bt + (size_t)n * ldb + koff;
  v8bf lo = *reinterpret_cast<const v8bf*>(p);
  v8bf hi = *reinterpret_cast<const v8bf*>(p + 16);
  return CAT16(lo, hi);
}

// ---------------- conversion kernels ----------------
__global__ void cvt_x_f32_bf16(const float* __restrict__ x, bf16* __restrict__ xb, int n) {
  int i = blockIdx.x * blockDim.x + threadIdx.x;
  if (i < n) xb[i] = (bf16)x[i];
}

// Transpose-convert the four 1024x1024 weights into K-major bf16: Wt[n][k] = W[k][n].
__global__ void cvt_w_t_bf16(const float* __restrict__ wq, const float* __restrict__ wk,
                             const float* __restrict__ wv, const float* __restrict__ wo,
                             bf16* __restrict__ wt) {
  int i = blockIdx.x * blockDim.x + threadIdx.x;           // [0, 4*1M)
  if (i >= 4 * 1048576) return;
  int w = i >> 20;
  int r = i & 1048575;
  int k = r & 1023;
  int n = r >> 10;
  const float* W = (w == 0) ? wq : (w == 1) ? wk : (w == 2) ? wv : wo;
  wt[((size_t)w << 20) + ((size_t)n << 10) + k] = (bf16)W[((size_t)k << 10) + n];
}

// ---------------- GEMM + fused epilogues ----------------
// M=8192 (=B*S), N=1024, K=1024. Block: 256 thr = 8 waves along M; wave tile 16x64.
// MODE 0: Q projection + RoPE -> Qh[B,H,S,64]   (bf16)
// MODE 1: K projection + RoPE -> Kh[B,H,S,64]   (bf16)
// MODE 2: V projection        -> Vt[B,H,64,S]   (bf16, transposed for P@V)
// MODE 3: out projection + bias -> fp32 [B,S,D]
template <int MODE>
__global__ void gemm_qkvo(const bf16* __restrict__ A, const bf16* __restrict__ Bt,
                          const float* __restrict__ bias,
                          bf16* __restrict__ outb, float* __restrict__ outf) {
  const int lane = threadIdx.x & 31;
  const int wave = threadIdx.x >> 5;
  const int m0 = blockIdx.x * 128 + wave * 16;
  const int n0 = blockIdx.y * 64;                 // one full head slice per wave

  v8f acc[4] = {};
  for (int k0 = 0; k0 < 1024; k0 += 32) {
    v16bf a = load_a_frag(A, 1024, m0, k0, lane);
#pragma unroll
    for (int f = 0; f < 4; ++f) {
      v16bf b = load_b_frag(Bt, 1024, n0 + f * 16, k0, lane);
      acc[f] = wmma_bf16(a, b, acc[f]);
    }
  }

  const int colbase = lane & 15;
  const int rowhalf = ((lane >> 4) & 1) * 8;      // C/D frag: M = r + rowhalf

  if (MODE == 3) {
#pragma unroll
    for (int f = 0; f < 4; ++f) {
      int n = n0 + f * 16 + colbase;
      float bv = bias[n];
#pragma unroll
      for (int r = 0; r < 8; ++r) {
        int m = m0 + r + rowhalf;
        outf[(size_t)m * 1024 + n] = acc[f][r] + bv;
      }
    }
  } else if (MODE == 2) {
#pragma unroll
    for (int f = 0; f < 4; ++f) {
      int n = n0 + f * 16 + colbase;
      int h = n >> 6, d = n & 63;
#pragma unroll
      for (int r = 0; r < 8; ++r) {
        int m = m0 + r + rowhalf;
        int b = m >> 11, s = m & 2047;
        outb[(((size_t)(b * 16 + h) * 64 + d) << 11) + s] = (bf16)acc[f][r];
      }
    }
  } else {
    // RoPE: out[j] = (x_j - x_{j+32})*sin(pos*theta_j); out[j+32] = (x_{j+32} + x_j)*sin(...)
    // n0 is 64-aligned -> frag f and f+2 hold the (j, j+32) pair in the same lane/reg.
#pragma unroll
    for (int f = 0; f < 2; ++f) {
      int n = n0 + f * 16 + colbase;
      int h = n >> 6;
      int j = n & 63;                                        // j in [0,32)
      float theta = exp2f(-(float)j * 0.83048202f);          // 10000^(-2j/32)
#pragma unroll
      for (int r = 0; r < 8; ++r) {
        int m = m0 + r + rowhalf;
        int b = m >> 11, s = m & 2047;
        float sv = __sinf((float)s * theta);
        float xr = acc[f][r], xi = acc[f + 2][r];
        size_t base = ((size_t)(b * 16 + h) * 2048 + s) * 64;
        outb[base + j]      = (bf16)((xr - xi) * sv);
        outb[base + j + 32] = (bf16)((xi + xr) * sv);
      }
    }
  }
}

// ---------------- flash attention (causal, online softmax) ----------------
// Block: 128 thr = 4 waves; each wave owns a 16-query tile of one (b,h).
// Qh,Kh: [B*H, S, 64] bf16; Vt: [B*H, 64, S] bf16; Y: [B, S, D] bf16.
__global__ void flash_attn(const bf16* __restrict__ Qh, const bf16* __restrict__ Kh,
                           const bf16* __restrict__ Vt, bf16* __restrict__ Y) {
  __shared__ bf16 pbuf[4][16 * 32];
  const int lane = threadIdx.x & 31;
  const int wave = threadIdx.x >> 5;
  const int q0 = blockIdx.x * 64 + wave * 16;
  const int bh = blockIdx.y;
  const bf16* Qb = Qh + (size_t)bh * 2048 * 64;
  const bf16* Kb = Kh + (size_t)bh * 2048 * 64;
  const bf16* Vb = Vt + (size_t)bh * 64 * 2048;
  bf16* pb = pbuf[wave];

  const int colbase = lane & 15;
  const int rowhalf = ((lane >> 4) & 1) * 8;

  v16bf qf0 = load_a_frag(Qb, 64, q0, 0, lane);
  v16bf qf1 = load_a_frag(Qb, 64, q0, 32, lane);

  v8f o[4] = {};
  float mrow[8], lrow[8];
#pragma unroll
  for (int r = 0; r < 8; ++r) { mrow[r] = -1e30f; lrow[r] = 0.f; }

  for (int kb = 0; kb <= q0 + 15; kb += 32) {
    // S = Q @ K^T for 32 keys (two 16-key fragments), K-dim = 64 in two steps.
    v8f sc0 = {}, sc1 = {};
    {
      v16bf b00 = load_b_frag(Kb, 64, kb, 0, lane);
      v16bf b01 = load_b_frag(Kb, 64, kb, 32, lane);
      sc0 = wmma_bf16(qf0, b00, sc0);
      sc0 = wmma_bf16(qf1, b01, sc0);
      v16bf b10 = load_b_frag(Kb, 64, kb + 16, 0, lane);
      v16bf b11 = load_b_frag(Kb, 64, kb + 16, 32, lane);
      sc1 = wmma_bf16(qf0, b10, sc1);
      sc1 = wmma_bf16(qf1, b11, sc1);
    }
    // fp32 online softmax on the C/D fragment layout (row = r + rowhalf);
    // 16-lane row reductions done entirely in DPP16 VALU (no DS waits).
#pragma unroll
    for (int r = 0; r < 8; ++r) {
      int qrow = q0 + r + rowhalf;
      float s0 = sc0[r] * 0.125f;                     // 1/sqrt(64)
      float s1 = sc1[r] * 0.125f;
      if (kb + colbase > qrow)      s0 = -1e30f;      // causal mask
      if (kb + 16 + colbase > qrow) s1 = -1e30f;
      float mnew = fmaxf(mrow[r], redmax16(fmaxf(s0, s1)));
      float corr = __expf(mrow[r] - mnew);
      mrow[r] = mnew;
      float p0 = __expf(s0 - mnew);
      float p1 = __expf(s1 - mnew);
      lrow[r] = lrow[r] * corr + redsum16(p0 + p1);
      o[0][r] *= corr; o[1][r] *= corr; o[2][r] *= corr; o[3][r] *= corr;
      int row = r + rowhalf;
      pb[row * 32 + colbase]      = (bf16)p0;         // stage P for A-fragment
      pb[row * 32 + 16 + colbase] = (bf16)p1;
    }
    // Re-read P in A-fragment order (same-wave DS ops are in-order).
    {
      const bf16* pr = &pb[colbase * 32 + rowhalf];
      v8bf lo = *reinterpret_cast<const v8bf*>(pr);
      v8bf hi = *reinterpret_cast<const v8bf*>(pr + 16);
      v16bf pa = CAT16(lo, hi);
#pragma unroll
      for (int f = 0; f < 4; ++f) {
        v16bf vb = load_b_frag(Vb, 2048, f * 16, kb, lane);   // V^T: contiguous along keys
        o[f] = wmma_bf16(pa, vb, o[f]);
      }
    }
  }

  const int b = bh >> 4, h = bh & 15;
#pragma unroll
  for (int f = 0; f < 4; ++f) {
    int d = f * 16 + colbase;
#pragma unroll
    for (int r = 0; r < 8; ++r) {
      int s = q0 + r + rowhalf;
      Y[((size_t)(b * 2048 + s)) * 1024 + h * 64 + d] = (bf16)(o[f][r] / lrow[r]);
    }
  }
}

// ---------------- launcher ----------------
extern "C" void kernel_launch(void* const* d_in, const int* in_sizes, int n_in,
                              void* d_out, int out_size, void* d_ws, size_t ws_size,
                              hipStream_t stream) {
  const float* x  = (const float*)d_in[0];
  const float* wq = (const float*)d_in[1];
  const float* wk = (const float*)d_in[2];
  const float* wv = (const float*)d_in[3];
  const float* wo = (const float*)d_in[4];
  const float* bo = (const float*)d_in[5];
  float* out = (float*)d_out;

  // Workspace layout (bf16 elements): 88 MB total.
  bf16* ws = (bf16*)d_ws;
  bf16* xb = ws;                         // x bf16             [8192,1024]   16 MB
  bf16* wt = ws + ((size_t)8  << 20);    // 4x Wt (K-major)    [4,1024,1024]  8 MB
  bf16* Qh = ws + ((size_t)12 << 20);    // Q (roped)          [64,2048,64]  16 MB
  bf16* Kh = ws + ((size_t)20 << 20);    // K (roped)          [64,2048,64]  16 MB
  bf16* Vt = ws + ((size_t)28 << 20);    // V^T                [64,64,2048]  16 MB
  bf16* yb = ws + ((size_t)36 << 20);    // attention output   [8192,1024]   16 MB

  cvt_x_f32_bf16<<<32768, 256, 0, stream>>>(x, xb, 8 * 1024 * 1024);
  cvt_w_t_bf16<<<16384, 256, 0, stream>>>(wq, wk, wv, wo, wt);

  dim3 gg(64, 16);   // M/128 x N/64
  gemm_qkvo<0><<<gg, 256, 0, stream>>>(xb, wt,                       nullptr, Qh, nullptr);
  gemm_qkvo<1><<<gg, 256, 0, stream>>>(xb, wt + ((size_t)1 << 20),   nullptr, Kh, nullptr);
  gemm_qkvo<2><<<gg, 256, 0, stream>>>(xb, wt + ((size_t)2 << 20),   nullptr, Vt, nullptr);

  flash_attn<<<dim3(32, 64), 128, 0, stream>>>(Qh, Kh, Vt, yb);

  gemm_qkvo<3><<<gg, 256, 0, stream>>>(yb, wt + ((size_t)3 << 20),   bo, nullptr, out);
}
